// DispatchByVariable_32693291057743
// MI455X (gfx1250) — matrix operand order repaired and checked by reading, not compiled, not run on hardware
//
#include <hip/hip_runtime.h>

// ---------------------------------------------------------------------------
// MoE-style bucketized dispatch GEMM for gfx1250 (MI455X).
// out[t] = x[t] @ W[bin(x[t][0])],  x: [32768, 512] f32, W: [8, 512, 512] f32.
// Per-workgroup counting sort by bin -> bin-homogeneous 16-row WMMA groups
// (boundary groups masked per-lane), bf16 v_wmma_f32_16x16x32_bf16.
// One group per wave (32 waves/WG): acc = 64 VGPRs, no spills, branch-free
// inner loop.  W K-slices double-buffered in LDS (72B padded rows: conflict-
// free b64 ops); wave-uniform expert conditions scalarized via readfirstlane.
// ---------------------------------------------------------------------------

typedef __attribute__((ext_vector_type(16))) __bf16       v16bf;
typedef __attribute__((ext_vector_type(8)))  float        v8f;
typedef __attribute__((ext_vector_type(8)))  unsigned int v8u;

#define N_TOK  32768
#define DIMD   512
#define TILE   512            // tokens per workgroup
#define CHUNK  128            // output columns per workgroup
#define NB     (CHUNK / 16)   // 8 N-blocks
#define NKB    (DIMD / 32)    // 16 K-blocks of 32
#define NGRP   (TILE / 16)    // 32 row groups == 32 waves
#define NBINS  8
#define BLOCK  1024           // 32 waves (wave32), one WGP
#define LDS_STRIDE 36         // shorts per n-row (72B): conflict-free banks

__device__ __forceinline__ int rfl(int v) {
  return __builtin_amdgcn_readfirstlane(v);
}

__device__ __forceinline__ unsigned pk_bf16(float lo, float hi) {
#if __has_builtin(__builtin_amdgcn_cvt_pk_bf16_f32)
  auto r = __builtin_amdgcn_cvt_pk_bf16_f32(lo, hi);
  return __builtin_bit_cast(unsigned, r);
#else
  unsigned a = __float_as_uint(lo), b = __float_as_uint(hi);
  a = (a + 0x7FFFu + ((a >> 16) & 1u)) >> 16;   // RNE
  b = (b + 0x7FFFu + ((b >> 16) & 1u)) >> 16;
  return a | (b << 16);
#endif
}

// A fragment: 16x32 bf16, lane L(<16)=row M, K {0..7,16..23}; lane L+16 same
// row, K {8..15,24..31}.  Invalid (wrong-bin) rows are zeroed per-lane.
__device__ __forceinline__ v16bf load_a_frag(const float* __restrict__ xrow,
                                             int off, bool valid) {
  v8u u = {0u, 0u, 0u, 0u, 0u, 0u, 0u, 0u};
  if (valid) {
    const float4* p = (const float4*)(xrow + off);
    float4 f0 = p[0], f1 = p[1];
    const float4* q = (const float4*)(xrow + off + 16);
    float4 f2 = q[0], f3 = q[1];
    u[0] = pk_bf16(f0.x, f0.y); u[1] = pk_bf16(f0.z, f0.w);
    u[2] = pk_bf16(f1.x, f1.y); u[3] = pk_bf16(f1.z, f1.w);
    u[4] = pk_bf16(f2.x, f2.y); u[5] = pk_bf16(f2.z, f2.w);
    u[6] = pk_bf16(f3.x, f3.y); u[7] = pk_bf16(f3.z, f3.w);
  }
  return __builtin_bit_cast(v16bf, u);
}

// B fragment from padded LDS tile [n][d] bf16: lane L(<16)=col N, K {0..7,16..23}
// of the 32-block; lane L+16 = same col, K {8..15,24..31}.  ds_load_2addr_b64 pairs.
__device__ __forceinline__ v16bf load_b_frag(const unsigned short* base,
                                             int nb, int laneM, int laneH) {
  const unsigned short* s = base + (nb * 16 + laneM) * LDS_STRIDE + laneH * 8;
  uint2 g0 = *(const uint2*)(s + 0);
  uint2 g1 = *(const uint2*)(s + 4);
  uint2 g2 = *(const uint2*)(s + 16);
  uint2 g3 = *(const uint2*)(s + 20);
  v8u u;
  u[0] = g0.x; u[1] = g0.y; u[2] = g1.x; u[3] = g1.y;
  u[4] = g2.x; u[5] = g2.y; u[6] = g3.x; u[7] = g3.y;
  return __builtin_bit_cast(v16bf, u);
}

// Cooperative stage of W[k][kb*32..+31][cb..cb+127] -> LDS bf16 [n][d].
// 1024 threads: thread owns (n, d-quad): 4 coalesced global reads (per-d
// 128B rows), one contiguous b64 LDS store; 72B stride => conflict-free.
__device__ __forceinline__ void stage_w(const float* __restrict__ W,
                                        unsigned short* dst, int k, int kb,
                                        int cb, int n, int d0) {
  const float* col = W + ((size_t)k * DIMD + (size_t)(kb * 32 + d0)) * DIMD + cb + n;
  float v0 = col[0 * DIMD], v1 = col[1 * DIMD];
  float v2 = col[2 * DIMD], v3 = col[3 * DIMD];
  uint2 a;
  a.x = pk_bf16(v0, v1); a.y = pk_bf16(v2, v3);
  *(uint2*)(dst + n * LDS_STRIDE + d0) = a;
}

#define WMMA_BF16(a, b, c) \
  __builtin_amdgcn_wmma_f32_16x16x32_bf16(false, (a), false, (b), (short)0, (c), false, false)

__global__ __launch_bounds__(BLOCK, 1)
void moe_bucket_gemm(const float* __restrict__ x,
                     const float* __restrict__ W,
                     float* __restrict__ out) {
  __shared__ __align__(16) unsigned short ldsB[2][CHUNK * LDS_STRIDE];  // 2x9KB
  __shared__ short         s_perm[TILE];
  __shared__ unsigned char s_bin[TILE];
  __shared__ int           s_cnt[NBINS];
  __shared__ int           s_base[NBINS + 1];
  __shared__ unsigned char s_glo[NGRP], s_ghi[NGRP];

  const int    tid   = threadIdx.x;
  const size_t tbase = (size_t)blockIdx.x * TILE;
  const int    cb    = blockIdx.y * CHUNK;

  // ---- phase A: bucketize + counting sort (512 tokens, first 512 threads) --
  if (tid < NBINS) s_cnt[tid] = 0;
  __syncthreads();
  int bin = 0, rank = 0;
  if (tid < TILE) {
    float y = x[(tbase + tid) * DIMD];  // INDEX == 0
    bin = (y > -1.5f) + (y > -1.0f) + (y > -0.5f) + (y > 0.0f) +
          (y >  0.5f) + (y >  1.0f) + (y >  1.5f);
    s_bin[tid] = (unsigned char)bin;
    rank = atomicAdd(&s_cnt[bin], 1);
  }
  __syncthreads();
  if (tid == 0) {
    int b = 0;
    for (int i = 0; i < NBINS; ++i) { s_base[i] = b; b += s_cnt[i]; }
    s_base[NBINS] = b;
  }
  __syncthreads();
  if (tid < TILE) s_perm[s_base[bin] + rank] = (short)tid;
  __syncthreads();
  if (tid < NGRP) {  // bin run [lo,hi] per sorted 16-row group
    s_glo[tid] = s_bin[s_perm[tid * 16]];
    s_ghi[tid] = s_bin[s_perm[tid * 16 + 15]];
  }
  __syncthreads();

  // ---- phase B: grouped GEMM via WMMA (one 16-row group per wave) ----
  const int wave  = tid >> 5;            // == group id g
  const int lane  = tid & 31;
  const int laneM = lane & 15;
  const int laneH = (lane >> 4) & 1;
  const float* xtile = x + tbase * DIMD;

  const int g      = wave;
  const int row    = s_perm[g * 16 + laneM];
  const int mybin  = s_bin[row];
  const float* xr  = xtile + (size_t)row * DIMD;
  // bin run identical across the wave: force into SGPRs -> scalar branches.
  const int lo = rfl(s_glo[g]), hi = rfl(s_ghi[g]);

  v8f acc[NB];
#pragma unroll
  for (int nb = 0; nb < NB; ++nb)
#pragma unroll
    for (int r = 0; r < 8; ++r) acc[nb][r] = 0.0f;

  const int st_n  = tid & 127;        // column within chunk
  const int st_d0 = (tid >> 7) * 4;   // d-quad within 32-row K block

  for (int k = 0; k < NBINS; ++k) {
    if (rfl(s_cnt[k]) == 0) continue;          // uniform: skip empty experts
    const bool need = (k >= lo) && (k <= hi);  // scalar, wave-uniform
    // prologue: stage first K-slice into buffer 0 (buffer 1 was last consumed)
    stage_w(W, ldsB[0], k, 0, cb, st_n, st_d0);
    for (int kb = 0; kb < NKB; ++kb) {
      __syncthreads();  // staging of kb visible; compute of kb-1 finished
      if (kb + 1 < NKB)  // overlap: stage next slice while computing this one
        stage_w(W, ldsB[(kb + 1) & 1], k, kb + 1, cb, st_n, st_d0);
      if (!need) continue;  // barrier is at loop head: safe
      const unsigned short* bbase = ldsB[kb & 1];
      v16bf a = load_a_frag(xr, kb * 32 + laneH * 8, mybin == k);
#pragma unroll
      for (int nb = 0; nb < NB; ++nb) {
        v16bf b = load_b_frag(bbase, nb, laneM, laneH);
        acc[nb] = WMMA_BF16(a, b, acc[nb]);
      }
    }
  }

  // ---- write out: D layout lanes0-15 M=r, lanes16-31 M=r+8; 64B coalesced --
#pragma unroll
  for (int r = 0; r < 8; ++r) {
    const int M     = laneH ? (r + 8) : r;
    const int rowid = s_perm[g * 16 + M];
    float* o = out + (tbase + rowid) * DIMD + cb + laneM;
#pragma unroll
    for (int nb = 0; nb < NB; ++nb) o[nb * 16] = acc[nb][r];
  }
}

extern "C" void kernel_launch(void* const* d_in, const int* in_sizes, int n_in,
                              void* d_out, int out_size, void* d_ws, size_t ws_size,
                              hipStream_t stream) {
  (void)in_sizes; (void)n_in; (void)d_ws; (void)ws_size; (void)out_size;
  const float* x = (const float*)d_in[0];   // [8*4096, 512] f32
  const float* W = (const float*)d_in[1];   // [8, 512, 512] f32
  float*       o = (float*)d_out;           // [8*4096, 512] f32
  dim3 grid(N_TOK / TILE, DIMD / CHUNK);    // (64, 4)
  moe_bucket_gemm<<<grid, BLOCK, 0, stream>>>(x, W, o);
}
